// MyRnn_30099130810550
// MI455X (gfx1250) — compile-verified, matrix operand order checked by reading, not run on hardware
//
#include <hip/hip_runtime.h>
#include <math.h>

// ---------------------------------------------------------------------------
// Two-layer SimpleRNN, fused, WMMA-based (gfx1250 / MI455X).
//
// B=512, T=256, D=128, U=256.
//   h0 = tanh(x_t @ k0 + h0 @ rk0 + b0)
//   h1 = tanh(h0 @ k1 + h1 @ rk1 + b1)
//   out = sigmoid(h1_T @ wo + bo)
//
// 32 workgroups x 16 batch rows (rows independent -> no inter-WG sync).
// Each WG: 256 threads = 8 wave32s; wave w owns output cols [32w, 32w+32)
// via v_wmma_f32_16x16x32_bf16 (f32 accumulate). Recurrent state double-
// buffered bf16 in LDS; weights pre-packed bf16 in d_ws and held in the
// high VGPR file across the whole T-loop by the compiler.
// Two barriers/step; x_{t+1} prefetch overlapped with layer-1 compute.
// ---------------------------------------------------------------------------

#define B_ 512
#define T_ 256
#define D_ 128
#define U_ 256

typedef __attribute__((ext_vector_type(16))) __bf16 v16bf;
typedef __attribute__((ext_vector_type(8)))  __bf16 v8bf;
typedef __attribute__((ext_vector_type(8)))  float  v8f;

// Packed-weight element offsets in d_ws (bf16 elements).
#define WS_PK0   0        // k0 : 128x256 -> 64 frags * 512
#define WS_PRK0  32768    // rk0: 256x256 -> 128 frags * 512
#define WS_PK1   98304    // k1
#define WS_PRK1  163840   // rk1   (total 229376 bf16 = 448 KB)

// A-fragment (16x32 MxK) from row-major LDS [16][rowStride] per ISA 7.12.2:
// lane l (m=l&15, kh=l>>4) holds K = {kc*32+kh*8+0..7, kc*32+16+kh*8+0..7}.
static __device__ __forceinline__ v16bf
load_a_frag(const __bf16* base, int rowStride, int kc, int lane) {
    const int m  = lane & 15;
    const int kh = lane >> 4;
    const __bf16* p = base + m * rowStride + kc * 32 + kh * 8;
    v8bf lo = *(const v8bf*)p;
    v8bf hi = *(const v8bf*)(p + 16);
    return __builtin_shufflevector(lo, hi,
        0, 1, 2, 3, 4, 5, 6, 7, 8, 9, 10, 11, 12, 13, 14, 15);
}

// B-fragment: packed so fragment f, lane l owns 16 contiguous bf16 (32 B).
static __device__ __forceinline__ v16bf
load_b_frag(const __bf16* pw, int frag, int lane) {
    return *(const v16bf*)(pw + frag * 512 + lane * 16);
}

static __device__ __forceinline__ v8f
wmma_bf16(v16bf a, v16bf b, v8f c) {
    return __builtin_amdgcn_wmma_f32_16x16x32_bf16(
        false, a, false, b, (short)0, c, false, false);
}

// Fast tanh: 1 - 2/(exp(2x)+1). v_exp_f32 (TRANS) co-executes with WMMA.
static __device__ __forceinline__ float fast_tanh(float x) {
    const float e = __expf(2.0f * x);
    return 1.0f - 2.0f * __builtin_amdgcn_rcpf(e + 1.0f);
}

// ---------------------------------------------------------------------------
// Pre-kernel: pack fp32 weights [K][N=256] -> bf16 B-fragment layout.
// Fragment (kc, nt) = 32x16 block; lane l (n = nt*16+(l&15), kh = l>>4)
// stores K-locals {kh*8+0..7, 16+kh*8+0..7} contiguously (mirror of A).
// Grid: 448 blocks x 32 lanes.
// ---------------------------------------------------------------------------
__global__ __launch_bounds__(32) void pack_weights_kernel(
    const float* __restrict__ k0, const float* __restrict__ rk0,
    const float* __restrict__ k1, const float* __restrict__ rk1,
    __bf16* __restrict__ pw) {
    const int f    = blockIdx.x;
    const int lane = threadIdx.x;

    const float* W;
    int base, lf;
    if (f < 64)        { W = k0;  base = WS_PK0;  lf = f;       }
    else if (f < 192)  { W = rk0; base = WS_PRK0; lf = f - 64;  }
    else if (f < 320)  { W = k1;  base = WS_PK1;  lf = f - 192; }
    else               { W = rk1; base = WS_PRK1; lf = f - 320; }

    const int kc = lf >> 4;
    const int nt = lf & 15;
    const int kh = lane >> 4;
    const int n  = nt * 16 + (lane & 15);

    __bf16* dst = pw + base + lf * 512 + lane * 16;
#pragma unroll
    for (int j = 0; j < 16; ++j) {
        const int kl = (j < 8) ? (kh * 8 + j) : (16 + kh * 8 + (j - 8));
        dst[j] = (__bf16)W[(kc * 32 + kl) * 256 + n];
    }
}

// ---------------------------------------------------------------------------
// Main fused RNN kernel. 32 blocks x 256 threads (8 waves).
// ---------------------------------------------------------------------------
__global__ __launch_bounds__(256) void rnn_fused_kernel(
    const int*   __restrict__ inputs,  // [B][T]
    const float* __restrict__ emb,     // [V][D]
    const float* __restrict__ b0, const float* __restrict__ b1,
    const float* __restrict__ wo, const float* __restrict__ bo,
    const __bf16* __restrict__ pk0, const __bf16* __restrict__ prk0,
    const __bf16* __restrict__ pk1, const __bf16* __restrict__ prk1,
    float* __restrict__ out) {

    __shared__ __align__(32) __bf16 sH0[2][16][U_];
    __shared__ __align__(32) __bf16 sH1[2][16][U_];
    __shared__ __align__(32) __bf16 sX[2][16][D_];   // double-buffered x_t
    __shared__ float sRed[16][16];

    const int tid     = threadIdx.x;
    const int lane    = tid & 31;
    const int wave    = tid >> 5;
    const int rowBase = blockIdx.x * 16;

    // h0 = h1 = 0
    for (int i = tid; i < 16 * U_; i += 256) {
        (&sH0[0][0][0])[i] = (__bf16)0.0f;
        (&sH1[0][0][0])[i] = (__bf16)0.0f;
    }

    const int nt0 = wave * 2;
    const int nt1 = nt0 + 1;
    const int c0  = nt0 * 16 + (lane & 15);
    const int c1  = c0 + 16;
    const float bias00 = b0[c0], bias01 = b0[c1];
    const float bias10 = b1[c0], bias11 = b1[c1];
    const int g = lane >> 4;              // C/D layout: VGPR i -> row i + 8*g

    // x staging mapping: thread -> (row, 8-elem segment)
    const int xRow = tid >> 4;
    const int xSeg = tid & 15;

    // Prologue: stage x_0 into sX[0].
    {
        const int idx = inputs[(rowBase + xRow) * T_ + 0];
        const float* src = emb + (long)idx * D_ + xSeg * 8;
#pragma unroll
        for (int j = 0; j < 8; ++j)
            sX[0][xRow][xSeg * 8 + j] = (__bf16)src[j];
    }
    __syncthreads();

    for (int t = 0; t < T_; ++t) {
        const int cur = t & 1;
        const int nxt = cur ^ 1;

        // Issue the x_{t+1} gather early; LDS store deferred to layer-1 phase.
        const int tn = (t + 1 < T_) ? (t + 1) : t;
        const int idxN = inputs[(rowBase + xRow) * T_ + tn];
        const float* srcN = emb + (long)idxN * D_ + xSeg * 8;
        float xf[8];
#pragma unroll
        for (int j = 0; j < 8; ++j) xf[j] = srcN[j];

        // ---- layer 0: h0' = tanh(x_t @ k0 + h0 @ rk0 + b0) ----
        {
            v8f acc0 = {bias00, bias00, bias00, bias00, bias00, bias00, bias00, bias00};
            v8f acc1 = {bias01, bias01, bias01, bias01, bias01, bias01, bias01, bias01};
#pragma unroll
            for (int kc = 0; kc < D_ / 32; ++kc) {          // x @ k0
                v16bf a = load_a_frag(&sX[cur][0][0], D_, kc, lane);
                acc0 = wmma_bf16(a, load_b_frag(pk0, kc * 16 + nt0, lane), acc0);
                acc1 = wmma_bf16(a, load_b_frag(pk0, kc * 16 + nt1, lane), acc1);
            }
#pragma unroll
            for (int kc = 0; kc < U_ / 32; ++kc) {          // h0 @ rk0
                v16bf a = load_a_frag(&sH0[cur][0][0], U_, kc, lane);
                acc0 = wmma_bf16(a, load_b_frag(prk0, kc * 16 + nt0, lane), acc0);
                acc1 = wmma_bf16(a, load_b_frag(prk0, kc * 16 + nt1, lane), acc1);
            }
#pragma unroll
            for (int i = 0; i < 8; ++i) {
                const int r = i + 8 * g;
                sH0[nxt][r][c0] = (__bf16)fast_tanh(acc0[i]);
                sH0[nxt][r][c1] = (__bf16)fast_tanh(acc1[i]);
            }
        }
        __syncthreads();

        // Stage prefetched x_{t+1} into the other buffer (ordered by the
        // end-of-step barrier against next step's layer-0 reads).
#pragma unroll
        for (int j = 0; j < 8; ++j)
            sX[nxt][xRow][xSeg * 8 + j] = (__bf16)xf[j];

        // ---- layer 1: h1' = tanh(h0' @ k1 + h1 @ rk1 + b1) ----
        {
            v8f acc0 = {bias10, bias10, bias10, bias10, bias10, bias10, bias10, bias10};
            v8f acc1 = {bias11, bias11, bias11, bias11, bias11, bias11, bias11, bias11};
#pragma unroll
            for (int kc = 0; kc < U_ / 32; ++kc) {          // h0' @ k1
                v16bf a = load_a_frag(&sH0[nxt][0][0], U_, kc, lane);
                acc0 = wmma_bf16(a, load_b_frag(pk1, kc * 16 + nt0, lane), acc0);
                acc1 = wmma_bf16(a, load_b_frag(pk1, kc * 16 + nt1, lane), acc1);
            }
#pragma unroll
            for (int kc = 0; kc < U_ / 32; ++kc) {          // h1 @ rk1
                v16bf a = load_a_frag(&sH1[cur][0][0], U_, kc, lane);
                acc0 = wmma_bf16(a, load_b_frag(prk1, kc * 16 + nt0, lane), acc0);
                acc1 = wmma_bf16(a, load_b_frag(prk1, kc * 16 + nt1, lane), acc1);
            }
#pragma unroll
            for (int i = 0; i < 8; ++i) {
                const int r = i + 8 * g;
                sH1[nxt][r][c0] = (__bf16)fast_tanh(acc0[i]);
                sH1[nxt][r][c1] = (__bf16)fast_tanh(acc1[i]);
            }
        }
        __syncthreads();
    }

    // ---- epilogue: out[row] = sigmoid(h1_T @ wo + bo) ----
    const int finBuf = ((T_ - 1) & 1) ^ 1;
    {
        const int row = tid >> 4;
        const int seg = tid & 15;
        float partial = 0.f;
#pragma unroll
        for (int j = 0; j < 16; ++j) {
            const int u = seg * 16 + j;
            partial += (float)sH1[finBuf][row][u] * wo[u];
        }
        sRed[row][seg] = partial;
    }
    __syncthreads();
    if (tid < 16) {
        float s = 0.f;
#pragma unroll
        for (int j = 0; j < 16; ++j) s += sRed[tid][j];
        s += bo[0];
        out[rowBase + tid] = 1.0f / (1.0f + __expf(-s));
    }
}

// ---------------------------------------------------------------------------
// Inputs: 0 inputs[i32], 1 emb, 2 k0, 3 rk0, 4 b0, 5 k1, 6 rk1, 7 b1,
//         8 wo, 9 bo.  d_ws: 448 KB packed bf16 weights.
// ---------------------------------------------------------------------------
extern "C" void kernel_launch(void* const* d_in, const int* in_sizes, int n_in,
                              void* d_out, int out_size, void* d_ws, size_t ws_size,
                              hipStream_t stream) {
    const int*   inputs = (const int*)  d_in[0];
    const float* emb    = (const float*)d_in[1];
    const float* k0     = (const float*)d_in[2];
    const float* rk0    = (const float*)d_in[3];
    const float* b0     = (const float*)d_in[4];
    const float* k1     = (const float*)d_in[5];
    const float* rk1    = (const float*)d_in[6];
    const float* b1     = (const float*)d_in[7];
    const float* wo     = (const float*)d_in[8];
    const float* bo     = (const float*)d_in[9];
    float*  out = (float*)d_out;
    __bf16* pw  = (__bf16*)d_ws;

    pack_weights_kernel<<<448, 32, 0, stream>>>(k0, rk0, k1, rk1, pw);
    rnn_fused_kernel<<<B_ / 16, 256, 0, stream>>>(
        inputs, emb, b0, b1, wo, bo,
        pw + WS_PK0, pw + WS_PRK0, pw + WS_PK1, pw + WS_PRK1, out);
}
